// TemplatePairStack_24412594111362
// MI455X (gfx1250) — matrix operand order repaired and checked by reading, not compiled, not run on hardware
//
#include <hip/hip_runtime.h>

typedef __attribute__((ext_vector_type(16))) __bf16 v16bf;
typedef __attribute__((ext_vector_type(8)))  float  v8f;

#define DEV __device__ __forceinline__

DEV v8f wmma_bf16(v16bf a, v16bf b, v8f c) {
  return __builtin_amdgcn_wmma_f32_16x16x32_bf16(false, a, false, b, (short)0, c, false, false);
}

// A operand (16x32, 16-bit) from row-major LDS [rows][stride]; K = kbase..kbase+31.
// Contiguous K order (same as ldsB): lanes 0-15 row M=l, K 0-15; lanes 16-31 row M=l-16, K 16-31.
DEV v16bf ldsA(const __bf16* base, int stride, int row0, int kbase) {
  int l = threadIdx.x & 31;
  const __bf16* p = base + (row0 + (l & 15)) * stride + kbase + ((l >> 4) << 4);
  v16bf r;
#pragma unroll
  for (int i = 0; i < 16; ++i) r[i] = p[i];
  return r;
}

// B operand (32x16, 16-bit) from N-major LDS [Nrows][stride] (K contiguous per N row).
// Contiguous K order (same as ldsA): lanes 0-15 col N=l, K 0-15; lanes 16-31 col N=l-16, K 16-31.
DEV v16bf ldsB(const __bf16* base, int stride, int n0, int kbase) {
  int l = threadIdx.x & 31;
  const __bf16* p = base + (n0 + (l & 15)) * stride + kbase + ((l >> 4) << 4);
  v16bf r;
#pragma unroll
  for (int i = 0; i < 16; ++i) r[i] = p[i];
  return r;
}

DEV float sigf(float x) { return 1.0f / (1.0f + __expf(-x)); }

// ---------------- LayerNorm (+ optional triangle bias) -> bf16 x workspace ---------
__global__ void k_ln(const float* __restrict__ t, const float* __restrict__ g,
                     const float* __restrict__ b, const float* __restrict__ tbw,
                     __bf16* __restrict__ xw, float* __restrict__ tbo, int transposed) {
  int bi = blockIdx.x, bb = bi >> 8, i = bi & 255;
  int j = threadIdx.x;
  long src = transposed ? (((long)(bb * 256 + j) * 256 + i) * 64)
                        : (((long)(bb * 256 + i) * 256 + j) * 64);
  float v[64];
  float mean = 0.f;
#pragma unroll
  for (int c = 0; c < 64; ++c) { v[c] = t[src + c]; mean += v[c]; }
  mean *= (1.f / 64.f);
  float var = 0.f;
#pragma unroll
  for (int c = 0; c < 64; ++c) { float d = v[c] - mean; var += d * d; }
  float rs = rsqrtf(var * (1.f / 64.f) + 1e-5f);
  long dst = ((long)(bb * 256 + i) * 256 + j) * 64;
#pragma unroll
  for (int c = 0; c < 64; ++c) {
    float xn = (v[c] - mean) * rs * g[c] + b[c];
    v[c] = xn;
    xw[dst + c] = (__bf16)xn;
  }
  if (tbw) {
    float a0 = 0, a1 = 0, a2 = 0, a3 = 0;
#pragma unroll
    for (int c = 0; c < 64; ++c) {
      a0 += v[c] * tbw[c * 4 + 0]; a1 += v[c] * tbw[c * 4 + 1];
      a2 += v[c] * tbw[c * 4 + 2]; a3 += v[c] * tbw[c * 4 + 3];
    }
    long o = ((long)(bb * 4) * 256 + i) * 256 + j;
    tbo[o] = a0; tbo[o + 65536] = a1; tbo[o + 2 * 65536] = a2; tbo[o + 3 * 65536] = a3;
  }
}

// ---------------- Triangle attention (one workgroup per (map,row)) -----------------
__global__ void __launch_bounds__(256)
k_att(float* __restrict__ t, const __bf16* __restrict__ xw,
      const float* __restrict__ tbias, const float* __restrict__ mask,
      const float* __restrict__ qw, const float* __restrict__ kw,
      const float* __restrict__ vw, const float* __restrict__ gw,
      const float* __restrict__ gb, const float* __restrict__ ow,
      const float* __restrict__ ob, int transposed) {
  __shared__ __bf16 xs[256 * 64];
  __shared__ __bf16 wq[64 * 64], wk[64 * 64], wv[64 * 64], wg[64 * 64];
  __shared__ __bf16 qs[256 * 32];   // stride 32, cols 16..31 stay zero (K padding)
  __shared__ __bf16 ks[256 * 32];   // stride 32, cols 16..31 stay zero
  __shared__ __bf16 vt[16 * 256], gs[256 * 16];
  __shared__ __bf16 og[256 * 64];
  __shared__ __bf16 stage[8][16 * 32];
  __shared__ float biasK[256];

  int bi = blockIdx.x, bb = bi >> 8, i = bi & 255;
  int tid = threadIdx.x, l = tid & 31, w = tid >> 5;

  for (int e = tid; e < 4096; e += 256) {
    int k = e >> 6, nn = e & 63;
    wq[nn * 64 + k] = (__bf16)qw[e];
    wk[nn * 64 + k] = (__bf16)kw[e];
    wv[nn * 64 + k] = (__bf16)vw[e];
    wg[nn * 64 + k] = (__bf16)gw[e];
  }
  {   // zero the padded q/k stores once (K=16..31 region must be 0)
    unsigned* qz = (unsigned*)qs;
    unsigned* kz = (unsigned*)ks;
    for (int e = tid; e < 4096; e += 256) { qz[e] = 0u; kz[e] = 0u; }
  }
  {
    long base = ((long)(bb * 256 + i) * 256) * 64;
    for (int e = tid; e < 256 * 64; e += 256) xs[e] = xw[base + e];
  }
  {
    int j = tid;
    long ma = transposed ? ((long)(bb * 256 + j) * 256 + i) : ((long)(bb * 256 + i) * 256 + j);
    biasK[j] = 1e9f * (mask[ma] - 1.f);
  }
  __syncthreads();

  const float* tb = tbias + (long)(bb * 4) * 65536;
  int n = l & 15, mofs = (l >> 4) * 8;

  for (int h = 0; h < 4; ++h) {
    // per-head q,k,v,g projections: one A operand feeds 4 WMMAs, branch-free stores
    float gbn = gb[h * 16 + n];
    for (int mt = w; mt < 16; mt += 8) {
      v8f aq = {}, ak = {}, av = {}, ag = {};
#pragma unroll
      for (int kk2 = 0; kk2 < 2; ++kk2) {
        v16bf a = ldsA(xs, 64, mt * 16, kk2 * 32);
        aq = wmma_bf16(a, ldsB(wq, 64, h * 16, kk2 * 32), aq);
        ak = wmma_bf16(a, ldsB(wk, 64, h * 16, kk2 * 32), ak);
        av = wmma_bf16(a, ldsB(wv, 64, h * 16, kk2 * 32), av);
        ag = wmma_bf16(a, ldsB(wg, 64, h * 16, kk2 * 32), ag);
      }
#pragma unroll
      for (int r = 0; r < 8; ++r) {
        int m = mt * 16 + r + mofs;
        qs[m * 32 + n] = (__bf16)aq[r];
        ks[m * 32 + n] = (__bf16)ak[r];
        vt[n * 256 + m] = (__bf16)av[r];
        gs[m * 16 + n] = (__bf16)sigf(ag[r] + gbn);
      }
    }
    __syncthreads();

    // attention over 16 q-tiles, 2 per wave
    for (int jj = 0; jj < 2; ++jj) {
      int qt = w * 2 + jj;
      v16bf aqop = ldsA(qs, 32, qt * 16, 0);
      v8f sacc[16];
#pragma unroll
      for (int kt = 0; kt < 16; ++kt) {
        v8f z = {};
        sacc[kt] = wmma_bf16(aqop, ldsB(ks, 32, kt * 16, 0), z);
      }
      float mx[8];
#pragma unroll
      for (int r = 0; r < 8; ++r) mx[r] = -1e30f;
#pragma unroll
      for (int kt = 0; kt < 16; ++kt)
#pragma unroll
        for (int r = 0; r < 8; ++r) {
          int q = qt * 16 + r + mofs, kc = kt * 16 + n;
          float val = sacc[kt][r] * 0.25f + biasK[kc] + tb[((long)h * 256 + q) * 256 + kc];
          sacc[kt][r] = val;
          mx[r] = fmaxf(mx[r], val);
        }
#pragma unroll
      for (int r = 0; r < 8; ++r)
#pragma unroll
        for (int s = 1; s < 16; s <<= 1) mx[r] = fmaxf(mx[r], __shfl_xor(mx[r], s, 32));
      float sm[8];
#pragma unroll
      for (int r = 0; r < 8; ++r) sm[r] = 0.f;
#pragma unroll
      for (int kt = 0; kt < 16; ++kt)
#pragma unroll
        for (int r = 0; r < 8; ++r) {
          float e = __expf(sacc[kt][r] - mx[r]);
          sacc[kt][r] = e; sm[r] += e;
        }
#pragma unroll
      for (int r = 0; r < 8; ++r) {
#pragma unroll
        for (int s = 1; s < 16; s <<= 1) sm[r] += __shfl_xor(sm[r], s, 32);
        sm[r] = 1.f / sm[r];
      }
      v8f oacc = {};
#pragma unroll
      for (int kc = 0; kc < 8; ++kc) {
#pragma unroll
        for (int kh = 0; kh < 2; ++kh) {
          int kt = kc * 2 + kh;
#pragma unroll
          for (int r = 0; r < 8; ++r)
            stage[w][(r + mofs) * 32 + kh * 16 + n] = (__bf16)(sacc[kt][r] * sm[r]);
        }
        v16bf pa = ldsA(stage[w], 32, 0, 0);
        oacc = wmma_bf16(pa, ldsB(vt, 256, 0, kc * 32), oacc);
      }
#pragma unroll
      for (int r = 0; r < 8; ++r) {
        int q = qt * 16 + r + mofs;
        og[q * 64 + h * 16 + n] = (__bf16)(oacc[r] * (float)gs[q * 16 + n]);
      }
    }
    __syncthreads();
  }

  // output projection og @ o_w + o_b, residual into t
  for (int e = tid; e < 4096; e += 256) { int k = e >> 6, nn = e & 63; wq[nn * 64 + k] = (__bf16)ow[e]; }
  __syncthreads();
  for (int mt = w; mt < 16; mt += 8) {
    v8f acc[4];
    v8f z = {};
#pragma unroll
    for (int q = 0; q < 4; ++q) acc[q] = z;
#pragma unroll
    for (int kk2 = 0; kk2 < 2; ++kk2) {
      v16bf a = ldsA(og, 64, mt * 16, kk2 * 32);
#pragma unroll
      for (int nt = 0; nt < 4; ++nt)
        acc[nt] = wmma_bf16(a, ldsB(wq, 64, nt * 16, kk2 * 32), acc[nt]);
    }
#pragma unroll
    for (int nt = 0; nt < 4; ++nt)
#pragma unroll
      for (int r = 0; r < 8; ++r) {
        int m = mt * 16 + r + mofs, c = nt * 16 + n;
        long ta = transposed ? (((long)(bb * 256 + m) * 256 + i) * 64 + c)
                             : (((long)(bb * 256 + i) * 256 + m) * 64 + c);
        t[ta] += acc[nt][r] + ob[c];
      }
  }
}

// ---------------- Triangle-mult prep: a,b,g projections -----------------------------
__global__ void __launch_bounds__(256)
k_mulprep(const __bf16* __restrict__ xw, const float* __restrict__ mask,
          const float* agw, const float* agb, const float* apw, const float* apb,
          const float* bgw, const float* bgb, const float* bpw, const float* bpb,
          const float* ggw, const float* ggb,
          __bf16* __restrict__ aout, __bf16* __restrict__ bout,
          __bf16* __restrict__ gout, int incoming) {
  __shared__ __bf16 xs[256 * 64];
  __shared__ __bf16 wag[64 * 64], wap[64 * 64], wbg[64 * 64], wbp[64 * 64], wgg[64 * 64];
  __shared__ float mrow[256];
  int bi = blockIdx.x, bb = bi >> 8, i = bi & 255;
  int tid = threadIdx.x, l = tid & 31, w = tid >> 5;
  for (int e = tid; e < 4096; e += 256) {
    int k = e >> 6, nn = e & 63;
    wag[nn * 64 + k] = (__bf16)agw[e]; wap[nn * 64 + k] = (__bf16)apw[e];
    wbg[nn * 64 + k] = (__bf16)bgw[e]; wbp[nn * 64 + k] = (__bf16)bpw[e];
    wgg[nn * 64 + k] = (__bf16)ggw[e];
  }
  {
    long base = ((long)(bb * 256 + i) * 256) * 64;
    for (int e = tid; e < 256 * 64; e += 256) xs[e] = xw[base + e];
  }
  mrow[tid] = mask[(long)(bb * 256 + i) * 256 + tid];
  __syncthreads();
  int n = l & 15, mofs = (l >> 4) * 8;
  v8f z = {};

  // a-gates (sig(x@agw+agb)*(x@apw+apb)*mask) -> channel-major scatter
  for (int mt = w; mt < 16; mt += 8) {
    v8f gacc[4], pacc[4];
#pragma unroll
    for (int q = 0; q < 4; ++q) { gacc[q] = z; pacc[q] = z; }
#pragma unroll
    for (int kk2 = 0; kk2 < 2; ++kk2) {
      v16bf a = ldsA(xs, 64, mt * 16, kk2 * 32);
#pragma unroll
      for (int nt = 0; nt < 4; ++nt) {
        gacc[nt] = wmma_bf16(a, ldsB(wag, 64, nt * 16, kk2 * 32), gacc[nt]);
        pacc[nt] = wmma_bf16(a, ldsB(wap, 64, nt * 16, kk2 * 32), pacc[nt]);
      }
    }
#pragma unroll
    for (int nt = 0; nt < 4; ++nt)
#pragma unroll
      for (int r = 0; r < 8; ++r) {
        int m = mt * 16 + r + mofs, c = nt * 16 + n;
        float val = sigf(gacc[nt][r] + agb[c]) * (pacc[nt][r] + apb[c]) * mrow[m];
        long o = incoming ? (((long)(bb * 64 + c) * 256 + m) * 256 + i)
                          : (((long)(bb * 64 + c) * 256 + i) * 256 + m);
        aout[o] = (__bf16)val;
      }
  }
  // b-gates
  for (int mt = w; mt < 16; mt += 8) {
    v8f gacc[4], pacc[4];
#pragma unroll
    for (int q = 0; q < 4; ++q) { gacc[q] = z; pacc[q] = z; }
#pragma unroll
    for (int kk2 = 0; kk2 < 2; ++kk2) {
      v16bf a = ldsA(xs, 64, mt * 16, kk2 * 32);
#pragma unroll
      for (int nt = 0; nt < 4; ++nt) {
        gacc[nt] = wmma_bf16(a, ldsB(wbg, 64, nt * 16, kk2 * 32), gacc[nt]);
        pacc[nt] = wmma_bf16(a, ldsB(wbp, 64, nt * 16, kk2 * 32), pacc[nt]);
      }
    }
#pragma unroll
    for (int nt = 0; nt < 4; ++nt)
#pragma unroll
      for (int r = 0; r < 8; ++r) {
        int m = mt * 16 + r + mofs, c = nt * 16 + n;
        float val = sigf(gacc[nt][r] + bgb[c]) * (pacc[nt][r] + bpb[c]) * mrow[m];
        long o = incoming ? (((long)(bb * 64 + c) * 256 + m) * 256 + i)
                          : (((long)(bb * 64 + c) * 256 + i) * 256 + m);
        bout[o] = (__bf16)val;
      }
  }
  // output gate g = sigmoid(x@g_w + g_b), natural layout
  for (int mt = w; mt < 16; mt += 8) {
    v8f gacc[4];
#pragma unroll
    for (int q = 0; q < 4; ++q) gacc[q] = z;
#pragma unroll
    for (int kk2 = 0; kk2 < 2; ++kk2) {
      v16bf a = ldsA(xs, 64, mt * 16, kk2 * 32);
#pragma unroll
      for (int nt = 0; nt < 4; ++nt)
        gacc[nt] = wmma_bf16(a, ldsB(wgg, 64, nt * 16, kk2 * 32), gacc[nt]);
    }
#pragma unroll
    for (int nt = 0; nt < 4; ++nt)
#pragma unroll
      for (int r = 0; r < 8; ++r) {
        int m = mt * 16 + r + mofs, c = nt * 16 + n;
        gout[((long)(bb * 256 + i) * 256 + m) * 64 + c] = (__bf16)sigf(gacc[nt][r] + ggb[c]);
      }
  }
}

// ---------------- 128 batched 256^3 NT GEMMs (triangle einsum) ----------------------
__global__ void __launch_bounds__(256)
k_tri_gemm(const __bf16* __restrict__ aw, const __bf16* __restrict__ bw,
           float* __restrict__ ow2) {
  __shared__ __bf16 As[64 * 256], Bs[64 * 256];
  int x = blockIdx.x;
  int jt = x & 3, it = (x >> 2) & 3, c = (x >> 4) & 63, bb = x >> 10;
  int tid = threadIdx.x, l = tid & 31, w = tid >> 5;
  long abase = ((long)(bb * 64 + c) * 256 + it * 64) * 256;
  long bbase = ((long)(bb * 64 + c) * 256 + jt * 64) * 256;
  for (int e = tid; e < 64 * 256; e += 256) { As[e] = aw[abase + e]; Bs[e] = bw[bbase + e]; }
  __syncthreads();
  int n = l & 15, mofs = (l >> 4) * 8;
  int mt = w >> 1, nt0 = (w & 1) * 2;     // wave -> (mt, 2 consecutive nt): A reuse + 2-deep WMMA pipe
  v8f acc0 = {}, acc1 = {};
#pragma unroll
  for (int kk2 = 0; kk2 < 8; ++kk2) {
    v16bf a = ldsA(As, 256, mt * 16, kk2 * 32);
    acc0 = wmma_bf16(a, ldsB(Bs, 256, (nt0 + 0) * 16, kk2 * 32), acc0);
    acc1 = wmma_bf16(a, ldsB(Bs, 256, (nt0 + 1) * 16, kk2 * 32), acc1);
  }
#pragma unroll
  for (int r = 0; r < 8; ++r) {
    int gi = it * 64 + mt * 16 + r + mofs;
    int gj0 = jt * 64 + (nt0 + 0) * 16 + n;
    int gj1 = jt * 64 + (nt0 + 1) * 16 + n;
    ow2[((long)(bb * 256 + gi) * 256 + gj0) * 64 + c] = acc0[r];
    ow2[((long)(bb * 256 + gi) * 256 + gj1) * 64 + c] = acc1[r];
  }
}

// ---------------- LN(o) @ z_w + z_b, sigmoid gate, residual -------------------------
__global__ void __launch_bounds__(256)
k_mulout(float* __restrict__ t, const float* __restrict__ ow2,
         const __bf16* __restrict__ gw2,
         const float* __restrict__ lng, const float* __restrict__ lnb,
         const float* __restrict__ zw, const float* __restrict__ zb) {
  __shared__ __bf16 os[256 * 64];
  __shared__ __bf16 wz[64 * 64];
  int bi = blockIdx.x, bb = bi >> 8, i = bi & 255;
  int tid = threadIdx.x, l = tid & 31, w = tid >> 5;
  for (int e = tid; e < 4096; e += 256) { int k = e >> 6, nn = e & 63; wz[nn * 64 + k] = (__bf16)zw[e]; }
  {
    int j = tid;
    long src = ((long)(bb * 256 + i) * 256 + j) * 64;
    float v[64]; float mean = 0;
#pragma unroll
    for (int c = 0; c < 64; ++c) { v[c] = ow2[src + c]; mean += v[c]; }
    mean *= (1.f / 64.f);
    float var = 0;
#pragma unroll
    for (int c = 0; c < 64; ++c) { float d = v[c] - mean; var += d * d; }
    float rs = rsqrtf(var * (1.f / 64.f) + 1e-5f);
#pragma unroll
    for (int c = 0; c < 64; ++c) os[j * 64 + c] = (__bf16)((v[c] - mean) * rs * lng[c] + lnb[c]);
  }
  __syncthreads();
  int n = l & 15, mofs = (l >> 4) * 8;
  v8f z = {};
  for (int mt = w; mt < 16; mt += 8) {
    v8f acc[4];
#pragma unroll
    for (int q = 0; q < 4; ++q) acc[q] = z;
#pragma unroll
    for (int kk2 = 0; kk2 < 2; ++kk2) {
      v16bf a = ldsA(os, 64, mt * 16, kk2 * 32);
#pragma unroll
      for (int nt = 0; nt < 4; ++nt)
        acc[nt] = wmma_bf16(a, ldsB(wz, 64, nt * 16, kk2 * 32), acc[nt]);
    }
#pragma unroll
    for (int nt = 0; nt < 4; ++nt)
#pragma unroll
      for (int r = 0; r < 8; ++r) {
        int m = mt * 16 + r + mofs, c = nt * 16 + n;
        long ta = ((long)(bb * 256 + i) * 256 + m) * 64 + c;
        t[ta] += (acc[nt][r] + zb[c]) * (float)gw2[ta];
      }
  }
}

// ---------------- Transition MLP 64 -> 128 (ReLU) -> 64, residual -------------------
__global__ void __launch_bounds__(256)
k_trans(float* __restrict__ t, const __bf16* __restrict__ xw,
        const float* __restrict__ mask,
        const float* __restrict__ w1, const float* __restrict__ b1,
        const float* __restrict__ w2, const float* __restrict__ b2) {
  __shared__ __bf16 xs[256 * 64];
  __shared__ __bf16 w1t[128 * 64];
  __shared__ __bf16 w2t[64 * 128];
  __shared__ __bf16 hs[256 * 128];
  __shared__ float mrow[256];
  int bi = blockIdx.x, bb = bi >> 8, i = bi & 255;
  int tid = threadIdx.x, l = tid & 31, w = tid >> 5;
  for (int e = tid; e < 8192; e += 256) { int k = e >> 7, nn = e & 127; w1t[nn * 64 + k] = (__bf16)w1[e]; }
  for (int e = tid; e < 8192; e += 256) { int k = e >> 6, nn = e & 63; w2t[nn * 128 + k] = (__bf16)w2[e]; }
  {
    long base = ((long)(bb * 256 + i) * 256) * 64;
    for (int e = tid; e < 256 * 64; e += 256) xs[e] = xw[base + e];
  }
  mrow[tid] = mask[(long)(bb * 256 + i) * 256 + tid];
  __syncthreads();
  int n = l & 15, mofs = (l >> 4) * 8;
  v8f z = {};
  // GEMM1: [256,64]@[64,128] + b1, ReLU -> hs ; A reused across 8 N-tiles
  for (int mt = w; mt < 16; mt += 8) {
    v8f acc[8];
#pragma unroll
    for (int q = 0; q < 8; ++q) acc[q] = z;
#pragma unroll
    for (int kk2 = 0; kk2 < 2; ++kk2) {
      v16bf a = ldsA(xs, 64, mt * 16, kk2 * 32);
#pragma unroll
      for (int nt = 0; nt < 8; ++nt)
        acc[nt] = wmma_bf16(a, ldsB(w1t, 64, nt * 16, kk2 * 32), acc[nt]);
    }
#pragma unroll
    for (int nt = 0; nt < 8; ++nt)
#pragma unroll
      for (int r = 0; r < 8; ++r) {
        int m = mt * 16 + r + mofs, c = nt * 16 + n;
        hs[m * 128 + c] = (__bf16)fmaxf(acc[nt][r] + b1[c], 0.f);
      }
  }
  __syncthreads();
  // GEMM2: [256,128]@[128,64] + b2, *mask, residual
  for (int mt = w; mt < 16; mt += 8) {
    v8f acc[4];
#pragma unroll
    for (int q = 0; q < 4; ++q) acc[q] = z;
#pragma unroll
    for (int kk2 = 0; kk2 < 4; ++kk2) {
      v16bf a = ldsA(hs, 128, mt * 16, kk2 * 32);
#pragma unroll
      for (int nt = 0; nt < 4; ++nt)
        acc[nt] = wmma_bf16(a, ldsB(w2t, 128, nt * 16, kk2 * 32), acc[nt]);
    }
#pragma unroll
    for (int nt = 0; nt < 4; ++nt)
#pragma unroll
      for (int r = 0; r < 8; ++r) {
        int m = mt * 16 + r + mofs, c = nt * 16 + n;
        t[((long)(bb * 256 + i) * 256 + m) * 64 + c] += (acc[nt][r] + b2[c]) * mrow[m];
      }
  }
}

// ---------------- Final LayerNorm to output ----------------------------------------
__global__ void k_lnout(const float* __restrict__ t, const float* __restrict__ g,
                        const float* __restrict__ b, float* __restrict__ out) {
  int bi = blockIdx.x, bb = bi >> 8, i = bi & 255;
  int j = threadIdx.x;
  long src = ((long)(bb * 256 + i) * 256 + j) * 64;
  float v[64]; float mean = 0;
#pragma unroll
  for (int c = 0; c < 64; ++c) { v[c] = t[src + c]; mean += v[c]; }
  mean *= (1.f / 64.f);
  float var = 0;
#pragma unroll
  for (int c = 0; c < 64; ++c) { float d = v[c] - mean; var += d * d; }
  float rs = rsqrtf(var * (1.f / 64.f) + 1e-5f);
#pragma unroll
  for (int c = 0; c < 64; ++c) out[src + c] = (v[c] - mean) * rs * g[c] + b[c];
}

extern "C" void kernel_launch(void* const* d_in, const int* in_sizes, int n_in,
                              void* d_out, int out_size, void* d_ws, size_t ws_size,
                              hipStream_t stream) {
  (void)in_sizes; (void)n_in; (void)out_size; (void)ws_size;
  const float* t_in = (const float*)d_in[0];
  const float* mask = (const float*)d_in[1];
  char* w = (char*)d_ws;
  float*  t_buf = (float*)(w + 0);            // 33554432 B
  __bf16* x_ws  = (__bf16*)(w + 33554432);    // 16777216 B
  float*  tb_ws = (float*)(w + 50331648);     //  8388608 B
  __bf16* a_ws  = (__bf16*)(w + 58720256);    // 16777216 B
  __bf16* b_ws  = (__bf16*)(w + 75497472);    // 16777216 B
  __bf16* g_ws  = (__bf16*)(w + 92274688);    // 16777216 B
  float*  o_ws  = (float*)(w + 109051904);    // 33554432 B

  hipMemcpyAsync(t_buf, t_in, 33554432, hipMemcpyDeviceToDevice, stream);

  dim3 blk(256), grd(512), grd2(2048);
  auto P = [&](int idx) { return (const float*)d_in[idx]; };

  for (int bk = 0; bk < 2; ++bk) {
    int base = 2 + 58 * bk;
    int AE = base + 0, AS = base + 10, MI = base + 20, MO = base + 36, TR = base + 52;
    // att keys (sorted): g_b,g_w,k_w,ln_b,ln_g,o_b,o_w,q_w,tb_w,v_w
    k_ln<<<grd, blk, 0, stream>>>(t_buf, P(AS + 4), P(AS + 3), P(AS + 8), x_ws, tb_ws, 0);
    k_att<<<grd, blk, 0, stream>>>(t_buf, x_ws, tb_ws, mask,
        P(AS + 7), P(AS + 2), P(AS + 9), P(AS + 1), P(AS + 0), P(AS + 6), P(AS + 5), 0);
    k_ln<<<grd, blk, 0, stream>>>(t_buf, P(AE + 4), P(AE + 3), P(AE + 8), x_ws, tb_ws, 1);
    k_att<<<grd, blk, 0, stream>>>(t_buf, x_ws, tb_ws, mask,
        P(AE + 7), P(AE + 2), P(AE + 9), P(AE + 1), P(AE + 0), P(AE + 6), P(AE + 5), 1);
    // mul keys (sorted): a_g_b,a_g_w,a_p_b,a_p_w,b_g_b,b_g_w,b_p_b,b_p_w,g_b,g_w,ln_b,ln_g,ln_o_b,ln_o_g,z_b,z_w
    k_ln<<<grd, blk, 0, stream>>>(t_buf, P(MO + 11), P(MO + 10), nullptr, x_ws, tb_ws, 0);
    k_mulprep<<<grd, blk, 0, stream>>>(x_ws, mask,
        P(MO + 1), P(MO + 0), P(MO + 3), P(MO + 2), P(MO + 5), P(MO + 4), P(MO + 7), P(MO + 6),
        P(MO + 9), P(MO + 8), a_ws, b_ws, g_ws, 0);
    k_tri_gemm<<<grd2, blk, 0, stream>>>(a_ws, b_ws, o_ws);
    k_mulout<<<grd, blk, 0, stream>>>(t_buf, o_ws, g_ws, P(MO + 13), P(MO + 12), P(MO + 15), P(MO + 14));
    k_ln<<<grd, blk, 0, stream>>>(t_buf, P(MI + 11), P(MI + 10), nullptr, x_ws, tb_ws, 0);
    k_mulprep<<<grd, blk, 0, stream>>>(x_ws, mask,
        P(MI + 1), P(MI + 0), P(MI + 3), P(MI + 2), P(MI + 5), P(MI + 4), P(MI + 7), P(MI + 6),
        P(MI + 9), P(MI + 8), a_ws, b_ws, g_ws, 1);
    k_tri_gemm<<<grd2, blk, 0, stream>>>(a_ws, b_ws, o_ws);
    k_mulout<<<grd, blk, 0, stream>>>(t_buf, o_ws, g_ws, P(MI + 13), P(MI + 12), P(MI + 15), P(MI + 14));
    // trans keys (sorted): b1,b2,ln_b,ln_g,w1,w2
    k_ln<<<grd, blk, 0, stream>>>(t_buf, P(TR + 3), P(TR + 2), nullptr, x_ws, tb_ws, 0);
    k_trans<<<grd, blk, 0, stream>>>(t_buf, x_ws, mask, P(TR + 4), P(TR + 0), P(TR + 5), P(TR + 1));
  }
  k_lnout<<<grd, blk, 0, stream>>>(t_buf, P(119), P(118), (float*)d_out);
}
